// RingLoss_5669356835816
// MI455X (gfx1250) — matrix operand order repaired and checked by reading, not compiled, not run on hardware
//
#include <hip/hip_runtime.h>
#include <hip/hip_bf16.h>

// ---------------- problem constants (fixed by reference setup) ----------------
#define NROW 4096
#define DDIM 512
#define TWO_N (2 * NROW)
#define TDROP 409          // int(N * 0.1)
#define NB 1024            // histogram bins per refinement pass
#define BLK 256
#define KSTEPS (DDIM / 32) // 16 k-steps of 32

typedef __attribute__((ext_vector_type(8)))  __bf16 v8bf;
typedef __attribute__((ext_vector_type(16))) __bf16 v16bf;
typedef __attribute__((ext_vector_type(8)))  float  v8f;

// ---------------- K1: row L2-normalize, f32 -> bf16 ----------------
__global__ void __launch_bounds__(BLK)
normalize_rows(const float* __restrict__ in, __bf16* __restrict__ out) {
    __shared__ float red[BLK];
    const int row = blockIdx.x;
    const float* r = in + (size_t)row * DDIM;
    float ss = 0.0f;
    for (int j = threadIdx.x; j < DDIM; j += BLK) { float v = r[j]; ss += v * v; }
    red[threadIdx.x] = ss; __syncthreads();
    for (int s = BLK / 2; s > 0; s >>= 1) {
        if (threadIdx.x < s) red[threadIdx.x] += red[threadIdx.x + s];
        __syncthreads();
    }
    const float inv = 1.0f / fmaxf(sqrtf(red[0]), 1e-12f);
    __bf16* o = out + (size_t)row * DDIM;
    for (int j = threadIdx.x; j < DDIM; j += BLK) o[j] = (__bf16)(r[j] * inv);
}

// ---------------- K2: S = X * Y^T via v_wmma_f32_16x16x32_bf16 ----------------
// Workgroup = 8 waves; wave w owns a 16(M) x 64(N) strip -> block tile 128 x 64.
// The 64x32 B tile (4 KB) is shared by all 8 waves: staged into LDS with
// double-buffered global_load_async_to_lds_b128 (ASYNCcnt) so the k+1 prefetch
// overlaps the k-step WMMAs. A fragments are per-wave-private -> direct loads.
// Grid: (NROW/128, NROW/64).
__global__ void __launch_bounds__(BLK)
gemm_nt_bf16(const __bf16* __restrict__ X, const __bf16* __restrict__ Y,
             float* __restrict__ S) {
    // [buf][c-subtile][lane-slot][16 bf16 fragment] ; 2 x 4 KB
    __shared__ __attribute__((aligned(32))) __bf16 sB[2][4][32][16];

    const int tid  = threadIdx.x;
    const int wave = tid >> 5;
    const int lane = tid & 31;
    const int m    = lane & 15;
    const int rowBase = blockIdx.x * 128 + wave * 16;
    const int colBase = blockIdx.y * 64;

    // ---- async B staging: each thread moves one 16B chunk per k-step ----
    const int bc    = tid >> 6;          // 0..3 : 16-column subtile
    const int bslot = (tid >> 1) & 31;   // lane slot within the fragment
    const int bhalf = tid & 1;           // low/high 16B of the 32B fragment
    // B fragment layout (16-bit B 32x16): lane = N; lanes 0-15 hold K {k0..+7, k0+16..+23},
    // lanes 16-31 hold +8 on both chunks (matches A).
    const int bcol  = colBase + bc * 16 + (bslot & 15);
    const int bkoff = ((bslot < 16) ? 0 : 8) + bhalf * 16;
    const __bf16* gB = Y + (size_t)bcol * DDIM + bkoff;
    const unsigned ldsB0 = (unsigned)(uintptr_t)&sB[0][bc][bslot][bhalf * 8];
    const unsigned ldsB1 = (unsigned)(uintptr_t)&sB[1][bc][bslot][bhalf * 8];

    // A fragment gather (16-bit A 16x32 layout): lane<16 -> K {k0..+7, k0+16..+23},
    // lane>=16 -> +8 on both chunks.
    const int koffA = (lane < 16) ? 0 : 8;
    const __bf16* xrow = X + (size_t)(rowBase + m) * DDIM + koffA;

    v8f acc0 = {}; v8f acc1 = {}; v8f acc2 = {}; v8f acc3 = {};

    // prologue: stage k0 = 0 into buffer 0
    asm volatile("global_load_async_to_lds_b128 %0, %1, off"
                 :: "v"(ldsB0), "v"(gB) : "memory");

#pragma unroll
    for (int j = 0; j < KSTEPS; ++j) {
        const int k0 = j * 32;
        if (j + 1 < KSTEPS) {
            const __bf16* gNext = gB + (k0 + 32);
            asm volatile("global_load_async_to_lds_b128 %0, %1, off"
                         :: "v"((j & 1) ? ldsB0 : ldsB1), "v"(gNext) : "memory");
            // own loads are in-order: <=1 outstanding => buffer j&1 landed
            asm volatile("s_wait_asynccnt 0x1" ::: "memory");
        } else {
            asm volatile("s_wait_asynccnt 0x0" ::: "memory");
        }
        __syncthreads();   // whole tile resident for all waves

        v8bf a0 = *(const v8bf*)(xrow + k0);
        v8bf a1 = *(const v8bf*)(xrow + k0 + 16);
        v16bf a = __builtin_shufflevector(a0, a1,
                    0,1,2,3,4,5,6,7,8,9,10,11,12,13,14,15);

        const __bf16* bbuf = &sB[j & 1][0][lane][0];
        v16bf b0 = *(const v16bf*)(bbuf + 0 * 512);
        v16bf b1 = *(const v16bf*)(bbuf + 1 * 512);
        v16bf b2 = *(const v16bf*)(bbuf + 2 * 512);
        v16bf b3 = *(const v16bf*)(bbuf + 3 * 512);

        acc0 = __builtin_amdgcn_wmma_f32_16x16x32_bf16(false, a, false, b0, (short)0, acc0, false, false);
        acc1 = __builtin_amdgcn_wmma_f32_16x16x32_bf16(false, a, false, b1, (short)0, acc1, false, false);
        acc2 = __builtin_amdgcn_wmma_f32_16x16x32_bf16(false, a, false, b2, (short)0, acc2, false, false);
        acc3 = __builtin_amdgcn_wmma_f32_16x16x32_bf16(false, a, false, b3, (short)0, acc3, false, false);

        __syncthreads();   // all consumes done before this buffer is re-staged
    }

    // 32-bit C/D layout: lane%16 = N, VGPR r -> M = r (lanes 0-15) / r+8 (lanes 16-31)
    const int rbase = rowBase + ((lane < 16) ? 0 : 8);
    const int cw    = colBase + m;
#pragma unroll
    for (int r = 0; r < 8; ++r) {
        const size_t ro = (size_t)(rbase + r) * NROW;
        S[ro + cw +  0] = acc0[r];
        S[ro + cw + 16] = acc1[r];
        S[ro + cw + 32] = acc2[r];
        S[ro + cw + 48] = acc3[r];
    }
}

// ---------------- K3 helpers: exact selection via histogram refinement ----------------
__device__ __forceinline__ float block_sum(float v, float* red) {
    const int tid = threadIdx.x;
    red[tid] = v; __syncthreads();
    for (int s = BLK / 2; s > 0; s >>= 1) {
        if (tid < s) red[tid] += red[tid + s];
        __syncthreads();
    }
    const float r = red[0];
    __syncthreads();
    return r;
}

// k-th smallest (1-indexed) of buf[0..n); returns bin-midpoint value (width < f32 ulp),
// bin lower edge lo, upper edge hi, rank-within-bin, and bin count.
__device__ void kth_refine(const float* __restrict__ buf, int n, int k,
                           int* hist, volatile int* shi,
                           float& vOut, float& loOut, float& hiOut,
                           int& kInBin, int& cntInBin) {
    const int tid = threadIdx.x;
    float lo = -1.05f, hi = 1.05f;
    int kk = k, cib = 0;
    for (int pass = 0; pass < 3; ++pass) {
        for (int b = tid; b < NB; b += BLK) hist[b] = 0;
        __syncthreads();
        const float scale = (float)NB / (hi - lo);
        for (int j = tid; j < n; j += BLK) {
            const float v = buf[j];
            if (v >= lo && v < hi) {
                int b = (int)((v - lo) * scale);
                b = b < 0 ? 0 : (b > NB - 1 ? NB - 1 : b);
                atomicAdd(&hist[b], 1);
            }
        }
        __syncthreads();
        if (tid == 0) {
            int c = 0, sel = NB - 1;
            for (int b = 0; b < NB; ++b) {
                const int h = hist[b];
                if (c + h >= kk) { sel = b; break; }
                c += h;
            }
            shi[0] = sel; shi[1] = kk - c; shi[2] = hist[sel];
        }
        __syncthreads();
        const int sel = shi[0];
        kk  = shi[1];
        cib = shi[2];
        const float w = (hi - lo) / (float)NB;
        const float nlo = lo + w * (float)sel;
        const float nhi = (sel == NB - 1) ? hi : (lo + w * (float)(sel + 1));
        lo = nlo; hi = nhi;
        __syncthreads();
    }
    vOut = 0.5f * (lo + hi);
    loOut = lo; hiOut = hi;
    kInBin = kk; cntInBin = cib;
}

// log( sum over middle (n - 2t) order statistics of exp(v/tau) ), exact w.r.t. ties.
__device__ float middle_logsum(const float* __restrict__ buf, int n, int t, float invTau,
                               int* hist, volatile int* shi, float* red) {
    float vLo, lo1, hi1, vHi, lo2, hi2;
    int kin1, cib1, kin2, cib2;
    kth_refine(buf, n, t,         hist, shi, vLo, lo1, hi1, kin1, cib1);
    kth_refine(buf, n, n - t + 1, hist, shi, vHi, lo2, hi2, kin2, cib2);

    float sTot = 0.0f, sLow = 0.0f, sHigh = 0.0f;
    for (int j = threadIdx.x; j < n; j += BLK) {
        const float v = buf[j];
        const float e = __expf(v * invTau);
        sTot += e;
        if (v < lo1)  sLow  += e;
        if (v >= hi2) sHigh += e;
    }
    sTot  = block_sum(sTot, red);
    sLow  = block_sum(sLow, red);
    sHigh = block_sum(sHigh, red);

    // bottom tail: t - kin1 values strictly below the final bin; kin1 dropped inside it
    const float dropBot = sLow + (float)kin1 * __expf(vLo * invTau);
    // top tail: cib2 - kin2 + 1 values of the final bin fall in the top-t set
    const float dropTop = sHigh + (float)(cib2 - kin2 + 1) * __expf(vHi * invTau);
    return __logf(sTot - dropBot - dropTop);
}

// ---------------- K3: per-row loss ----------------
__global__ void __launch_bounds__(BLK)
row_loss(const float* __restrict__ s11, const float* __restrict__ s12,
         const float* __restrict__ s22, const float* __restrict__ tauPtr,
         float* __restrict__ rloss) {
    __shared__ float buf[TWO_N];   // 32 KB of the 320 KB WGP LDS
    __shared__ int   hist[NB];
    __shared__ int   shi[4];
    __shared__ float red[BLK];

    const int i   = blockIdx.x;
    const int tid = threadIdx.x;
    const float invTau = 1.0f / tauPtr[0];

    // neg1 = [s11[i,:], s12[i,:]]
    for (int j = tid; j < NROW; j += BLK) {
        buf[j]        = s11[(size_t)i * NROW + j];
        buf[NROW + j] = s12[(size_t)i * NROW + j];
    }
    __syncthreads();
    const float logn1 = middle_logsum(buf, TWO_N, TDROP, invTau, hist, shi, red);
    __syncthreads();

    // neg2 = [s12[:,i], s22[i,:]]  (column reads stay L2-resident)
    for (int j = tid; j < NROW; j += BLK) {
        buf[j]        = s12[(size_t)j * NROW + i];
        buf[NROW + j] = s22[(size_t)i * NROW + j];
    }
    __syncthreads();
    const float logn2 = middle_logsum(buf, TWO_N, TDROP, invTau, hist, shi, red);

    if (tid == 0) {
        const float logpos = s12[(size_t)i * NROW + i] * invTau;
        rloss[i] = 0.5f * (logn1 + logn2) - logpos;
    }
}

// ---------------- K4: mean over rows ----------------
__global__ void __launch_bounds__(BLK)
reduce_mean(const float* __restrict__ rloss, float* __restrict__ out) {
    __shared__ float red[BLK];
    float s = 0.0f;
    for (int j = threadIdx.x; j < NROW; j += BLK) s += rloss[j];
    red[threadIdx.x] = s; __syncthreads();
    for (int st = BLK / 2; st > 0; st >>= 1) {
        if (threadIdx.x < st) red[threadIdx.x] += red[threadIdx.x + st];
        __syncthreads();
    }
    if (threadIdx.x == 0) out[0] = red[0] * (1.0f / (float)NROW);
}

// ---------------- host launcher ----------------
extern "C" void kernel_launch(void* const* d_in, const int* in_sizes, int n_in,
                              void* d_out, int out_size, void* d_ws, size_t ws_size,
                              hipStream_t stream) {
    (void)in_sizes; (void)n_in; (void)out_size; (void)ws_size;
    const float* h1  = (const float*)d_in[0];
    const float* h2  = (const float*)d_in[1];
    // d_in[2] = y (int64) — unused by the reference computation
    const float* tau = (const float*)d_in[3];

    char* ws = (char*)d_ws;
    const size_t HN_BYTES = (size_t)NROW * DDIM * sizeof(__bf16);   // 4 MB
    const size_t S_ELEMS  = (size_t)NROW * NROW;                    // 16M f32 = 64 MB
    __bf16* h1n  = (__bf16*)ws;
    __bf16* h2n  = (__bf16*)(ws + HN_BYTES);
    float*  s11  = (float*)(ws + 2 * HN_BYTES);
    float*  s12  = s11 + S_ELEMS;
    float*  s22  = s12 + S_ELEMS;
    float*  rls  = s22 + S_ELEMS;

    normalize_rows<<<NROW, BLK, 0, stream>>>(h1, h1n);
    normalize_rows<<<NROW, BLK, 0, stream>>>(h2, h2n);

    dim3 gg(NROW / 128, NROW / 64);
    gemm_nt_bf16<<<gg, BLK, 0, stream>>>(h1n, h1n, s11);
    gemm_nt_bf16<<<gg, BLK, 0, stream>>>(h1n, h2n, s12);
    gemm_nt_bf16<<<gg, BLK, 0, stream>>>(h2n, h2n, s22);

    row_loss<<<NROW, BLK, 0, stream>>>(s11, s12, s22, tau, rls);
    reduce_mean<<<1, BLK, 0, stream>>>(rls, (float*)d_out);
}
